// LoRAGAT_40578851013028
// MI455X (gfx1250) — matrix operand order repaired and checked by reading, not compiled, not run on hardware
//
#include <hip/hip_runtime.h>
#include <hip/hip_bf16.h>
#include <math.h>

typedef __attribute__((ext_vector_type(16))) _Float16 v16h;
typedef __attribute__((ext_vector_type(8)))  float    v8f;

#define IN_DIM 256
#define HEADS 8
#define CH 32
#define RANK 8
#define ATT_RANK 4
#define LORA_SCALE (32.0f / 8.0f)
#define ATT_SCALE  (32.0f / 4.0f)
#define NEG_SLOPE 0.2f

// ---------------- conversion / weight fusion ----------------

__global__ void k_cvt_f16(const float* __restrict__ in, _Float16* __restrict__ out, int n) {
  int i = blockIdx.x * blockDim.x + threadIdx.x;
  if (i < n) out[i] = (_Float16)in[i];
}

// Weff[o,k] = W[o,k] + LORA_SCALE * sum_r B[o,r]*A[r,k]   (O=K=256, R=8), stored f16
__global__ void k_fuse_w(const float* __restrict__ W, const float* __restrict__ A,
                         const float* __restrict__ B, _Float16* __restrict__ out) {
  int i = blockIdx.x * blockDim.x + threadIdx.x;   // 0..65535
  int o = i >> 8, k = i & 255;
  float acc = W[i];
#pragma unroll
  for (int r = 0; r < RANK; ++r)
    acc += LORA_SCALE * B[o * RANK + r] * A[r * IN_DIM + k];
  out[i] = (_Float16)acc;
}

// ---------------- WMMA GEMM: Y[M,256] = X[M,256] @ Weff[256,256]^T ----------------
// One 16x16 output tile per wave32; K loop of 8 x 32.
__global__ void k_gemm_wmma(const _Float16* __restrict__ X, const _Float16* __restrict__ Wt,
                            float* __restrict__ Y, int Mtiles) {
  const int NT = IN_DIM / 16;                       // 16 column tiles
  int wid  = threadIdx.x >> 5;
  int lane = threadIdx.x & 31;
  int tile = blockIdx.x * 8 + wid;
  if (tile >= Mtiles * NT) return;
  int mt = tile / NT, nt = tile % NT;

  const _Float16* xr = X  + (size_t)(mt * 16 + (lane & 15)) * IN_DIM;  // A: row = lane&15
  const _Float16* wr = Wt + (size_t)(nt * 16 + (lane & 15)) * IN_DIM;  // B: col = lane&15
  const int hiA = (lane >= 16) ? 8  : 0;
  const int hiB = (lane >= 16) ? 16 : 0;

  v8f acc = {};
#pragma unroll
  for (int kb = 0; kb < IN_DIM; kb += 32) {
    union { v16h v; uint32_t u[8]; } a, b;
#pragma unroll
    for (int j = 0; j < 8; ++j) {
      // A 16x32 f16 layout: lanes0-15 K={0..7,16..23}, lanes16-31 K={8..15,24..31}
      int ka = kb + ((j < 4) ? (2 * j) : (2 * j + 8)) + hiA;
      // B 32x16 f16 layout: lanes0-15 K={0..7,16..23}, lanes16-31 K={8..15,24..31}
      int kw = kb + ((j < 4) ? (2 * j) : (2 * j + 8)) + hiA;
      a.u[j] = *(const uint32_t*)(xr + ka);
      b.u[j] = *(const uint32_t*)(wr + kw);
    }
    acc = __builtin_amdgcn_wmma_f32_16x16x32_f16(false, a.v, false, b.v,
                                                 (short)0, acc, false, false);
  }
  // C/D layout: VGPR r -> M = r (+8 for lanes16-31), N = lane&15
  int ncol  = nt * 16 + (lane & 15);
  int mbase = mt * 16 + ((lane >= 16) ? 8 : 0);
#pragma unroll
  for (int r = 0; r < 8; ++r)
    Y[(size_t)(mbase + r) * IN_DIM + ncol] = acc[r];
}

// ---------------- attention ----------------

// eff[0..255]   = att_src + ATT_SCALE * Bs@As   (flattened [H,C])
// eff[256..511] = att_dst + ATT_SCALE * Bd@Ad
__global__ void k_att_eff(const float* __restrict__ atts, const float* __restrict__ As,
                          const float* __restrict__ Bs,   const float* __restrict__ attd,
                          const float* __restrict__ Ad,   const float* __restrict__ Bd,
                          float* __restrict__ eff) {
  int i = threadIdx.x;                               // 256 threads
  int c = i & 31;
  float l0 = 0.f, l1 = 0.f;
#pragma unroll
  for (int r = 0; r < ATT_RANK; ++r) {
    l0 += Bs[r] * As[r * CH + c];
    l1 += Bd[r] * Ad[r * CH + c];
  }
  eff[i]       = atts[i] + ATT_SCALE * l0;
  eff[256 + i] = attd[i] + ATT_SCALE * l1;
}

__global__ void k_scores(const float* __restrict__ xp, const float* __restrict__ eff,
                         float* __restrict__ ss, float* __restrict__ sd, int N) {
  int i = blockIdx.x * blockDim.x + threadIdx.x;
  if (i >= N * HEADS) return;
  int n = i >> 3, h = i & 7;
  const float* xr = xp + (size_t)n * IN_DIM + h * CH;
  float a0 = 0.f, a1 = 0.f;
#pragma unroll
  for (int c = 0; c < CH; ++c) {
    float v = xr[c];
    a0 += v * eff[h * CH + c];
    a1 += v * eff[256 + h * CH + c];
  }
  ss[i] = a0; sd[i] = a1;
}

// ---------------- edge phase ----------------

__global__ void k_init(float* __restrict__ agg, int* __restrict__ mx,
                       float* __restrict__ den, int N) {
  int i = blockIdx.x * blockDim.x + threadIdx.x;
  if (i < N * IN_DIM) agg[i] = 0.f;
  if (i < N * HEADS) { mx[i] = (int)0x80000000; den[i] = 0.f; }
}

__device__ __forceinline__ void edge_sd(const long long* __restrict__ ei, int E, int e,
                                        int& s, int& d) {
  if (e < E) { s = (int)ei[e]; d = (int)ei[E + e]; }
  else       { s = d = e - E; }                      // appended self-loops
}

__global__ void k_edge_alpha(const long long* __restrict__ ei, int E, int Etot,
                             const float* __restrict__ ss, const float* __restrict__ sd,
                             float* __restrict__ alpha, int* __restrict__ mx) {
  int i = blockIdx.x * blockDim.x + threadIdx.x;
  if (i >= Etot * HEADS) return;
  int e = i >> 3, h = i & 7;
  int s, d; edge_sd(ei, E, e, s, d);
  float a = ss[d * HEADS + h] + sd[s * HEADS + h];
  a = (a > 0.f) ? a : NEG_SLOPE * a;                 // leaky relu
  alpha[i] = a;
  int key = __float_as_int(a);
  key = (key >= 0) ? key : (key ^ 0x7FFFFFFF);       // order-preserving int map
  atomicMax(mx + d * HEADS + h, key);
}

__global__ void k_edge_exp(const long long* __restrict__ ei, int E, int Etot,
                           float* __restrict__ alpha, const int* __restrict__ mx,
                           float* __restrict__ den) {
  int i = blockIdx.x * blockDim.x + threadIdx.x;
  if (i >= Etot * HEADS) return;
  int e = i >> 3, h = i & 7;
  int s, d; edge_sd(ei, E, e, s, d);
  int key = mx[d * HEADS + h];
  float m = __int_as_float((key >= 0) ? key : (key ^ 0x7FFFFFFF));
  float ev = expf(alpha[i] - m);
  alpha[i] = ev;
  atomicAdd(den + d * HEADS + h, ev);
}

__global__ void k_edge_msg(const long long* __restrict__ ei, int E, int Etot,
                           const float* __restrict__ ev, const float* __restrict__ den,
                           const float* __restrict__ xp, float* __restrict__ agg) {
  int i = blockIdx.x * blockDim.x + threadIdx.x;     // Etot*256 threads
  if (i >= Etot * IN_DIM) return;
  int e = i >> 8, hc = i & 255, h = hc >> 5;
  int s, d; edge_sd(ei, E, e, s, d);
  float a = ev[e * HEADS + h] / den[d * HEADS + h];
  atomicAdd(agg + (size_t)d * IN_DIM + hc, xp[(size_t)s * IN_DIM + hc] * a);
}

// ---------------- epilogues ----------------

__global__ void k_elu_f16(const float* __restrict__ agg, _Float16* __restrict__ out, int n) {
  int i = blockIdx.x * blockDim.x + threadIdx.x;
  if (i >= n) return;
  float v = agg[i];
  float e = (v > 0.f) ? v : (expf(v) - 1.f);
  out[i] = (_Float16)e;
}

__global__ void k_mean(const float* __restrict__ agg, float* __restrict__ out, int N) {
  int i = blockIdx.x * blockDim.x + threadIdx.x;
  if (i >= N * CH) return;
  int n = i >> 5, c = i & 31;
  float s = 0.f;
#pragma unroll
  for (int h = 0; h < HEADS; ++h) s += agg[(size_t)n * IN_DIM + h * CH + c];
  out[i] = s * 0.125f;
}

// ---------------- host orchestration ----------------

extern "C" void kernel_launch(void* const* d_in, const int* in_sizes, int n_in,
                              void* d_out, int out_size, void* d_ws, size_t ws_size,
                              hipStream_t stream) {
  const float*     x  = (const float*)d_in[0];
  const long long* ei = (const long long*)d_in[1];
  const float* W [2] = {(const float*)d_in[2],  (const float*)d_in[11]};
  const float* A [2] = {(const float*)d_in[3],  (const float*)d_in[12]};
  const float* B [2] = {(const float*)d_in[4],  (const float*)d_in[13]};
  const float* AS[2] = {(const float*)d_in[5],  (const float*)d_in[14]};
  const float* AD[2] = {(const float*)d_in[6],  (const float*)d_in[15]};
  const float* LAs[2] = {(const float*)d_in[7],  (const float*)d_in[16]};
  const float* LBs[2] = {(const float*)d_in[8],  (const float*)d_in[17]};
  const float* LAd[2] = {(const float*)d_in[9],  (const float*)d_in[18]};
  const float* LBd[2] = {(const float*)d_in[10], (const float*)d_in[19]};

  const int N    = in_sizes[0] / IN_DIM;   // 10000
  const int E    = in_sizes[1] / 2;        // 160000
  const int Etot = E + N;                  // + self loops

  // workspace carve-up (256B aligned)
  char* ws = (char*)d_ws;
  size_t off = 0;
  auto carve = [&](size_t bytes) -> char* {
    char* p = ws + off;
    off = (off + bytes + 255) & ~(size_t)255;
    return p;
  };
  _Float16* xh    = (_Float16*)carve((size_t)N * IN_DIM * 2);     // x_f16, reused as elu(h)_f16
  _Float16* wh0   = (_Float16*)carve((size_t)IN_DIM * IN_DIM * 2);
  _Float16* wh1   = (_Float16*)carve((size_t)IN_DIM * IN_DIM * 2);
  float*    xp    = (float*)   carve((size_t)N * IN_DIM * 4);     // projected features (per layer)
  float*    agg   = (float*)   carve((size_t)N * IN_DIM * 4);     // aggregated messages
  float*    eff   = (float*)   carve(2 * 256 * 4);                // fused attention vectors
  float*    ssrc  = (float*)   carve((size_t)N * HEADS * 4);
  float*    sdst  = (float*)   carve((size_t)N * HEADS * 4);
  int*      mxb   = (int*)     carve((size_t)N * HEADS * 4);
  float*    den   = (float*)   carve((size_t)N * HEADS * 4);
  float*    alpha = (float*)   carve((size_t)Etot * HEADS * 4);
  (void)ws_size; (void)n_in; (void)out_size;

  const int TB = 256;
  auto blocks = [](long long n, int tb) { return (int)((n + tb - 1) / tb); };

  const int Mtiles = N / 16;                       // 625 (N is a multiple of 16)
  const int gemmBlocks = (Mtiles * (IN_DIM / 16) + 7) / 8;

  // prep
  k_cvt_f16<<<blocks((long long)N * IN_DIM, TB), TB, 0, stream>>>(x, xh, N * IN_DIM);
  k_fuse_w<<<blocks(IN_DIM * IN_DIM, TB), TB, 0, stream>>>(W[0], A[0], B[0], wh0);
  k_fuse_w<<<blocks(IN_DIM * IN_DIM, TB), TB, 0, stream>>>(W[1], A[1], B[1], wh1);

  for (int layer = 0; layer < 2; ++layer) {
    _Float16* wh = (layer == 0) ? wh0 : wh1;

    k_gemm_wmma<<<gemmBlocks, TB, 0, stream>>>(xh, wh, xp, Mtiles);
    k_att_eff<<<1, 256, 0, stream>>>(AS[layer], LAs[layer], LBs[layer],
                                     AD[layer], LAd[layer], LBd[layer], eff);
    k_scores<<<blocks((long long)N * HEADS, TB), TB, 0, stream>>>(xp, eff, ssrc, sdst, N);
    k_init<<<blocks((long long)N * IN_DIM, TB), TB, 0, stream>>>(agg, mxb, den, N);
    k_edge_alpha<<<blocks((long long)Etot * HEADS, TB), TB, 0, stream>>>(
        ei, E, Etot, ssrc, sdst, alpha, mxb);
    k_edge_exp<<<blocks((long long)Etot * HEADS, TB), TB, 0, stream>>>(
        ei, E, Etot, alpha, mxb, den);
    k_edge_msg<<<blocks((long long)Etot * IN_DIM, TB), TB, 0, stream>>>(
        ei, E, Etot, alpha, den, xp, agg);

    if (layer == 0) {
      k_elu_f16<<<blocks((long long)N * IN_DIM, TB), TB, 0, stream>>>(agg, xh, N * IN_DIM);
    } else {
      k_mean<<<blocks((long long)N * CH, TB), TB, 0, stream>>>(agg, (float*)d_out, N);
    }
  }
}
